// CG_30588757082611
// MI455X (gfx1250) — matrix operand order repaired and checked by reading, not compiled, not run on hardware
//
#include <hip/hip_runtime.h>
#include <hip/hip_bf16.h>
#include <stdint.h>

// ---------------------------------------------------------------------------
// GraphMAE-style loss on MI455X (gfx1250, wave32, WMMA).
//   * big GEMMs via v_wmma_f32_16x16x32_f16 (f16 in, f32 accum),
//     64x128 block tile, B fragments register-resident across row sub-tiles,
//     all LDS fragment traffic as 128-bit ds loads (padded, transposed W),
//     branch-free full-tile epilogue with float4 scale loads
//   * edge gather/scatter as float4 + f32 atomics (arrays are L2-resident)
//   * two-pass BN with block-reduced column stats
// ---------------------------------------------------------------------------

typedef __attribute__((ext_vector_type(16))) _Float16 v16h;
typedef __attribute__((ext_vector_type(8)))  _Float16 v8h;
typedef __attribute__((ext_vector_type(8)))  float    v8f;

static constexpr int DIM = 128;        // in_dim = hidden = out_dim
static constexpr int LDW = DIM + 8;    // padded LDS row stride (halves):
                                       // 272 B = 68 dwords -> 4-bank rotation

// ---------------------------------------------------------------------------
// Degree kernels: dn[i] = deg>0 ? deg^-1/2 : 0
// ---------------------------------------------------------------------------
__global__ void deg_count_kernel(const int* __restrict__ src, const int* __restrict__ dst,
                                 float* __restrict__ ds, float* __restrict__ dd, int ne) {
  int e = blockIdx.x * 256 + threadIdx.x;
  if (e >= ne) return;
  atomicAdd(&ds[src[e]], 1.0f);
  atomicAdd(&dd[dst[e]], 1.0f);
}

__global__ void deg_finish_kernel(float* __restrict__ ds, float* __restrict__ dd, int n) {
  int i = blockIdx.x * 256 + threadIdx.x;
  if (i >= n) return;
  float a = ds[i]; ds[i] = (a > 0.0f) ? rsqrtf(a) : 0.0f;
  float b = dd[i]; dd[i] = (b > 0.0f) ? rsqrtf(b) : 0.0f;
}

// ---------------------------------------------------------------------------
// WMMA GEMM: Y[m, :] = (X[m, :] @ W) * scale[m], optional masked rows of X
// treated as zero (the layer-1 "remainder") without mutating the input.
//
// Block = 256 threads = 8 waves; block tile = 64 rows x 128 cols.
// Wave w owns cols [16w,16w+16): it loads its four 16x32 B fragments from
// transposed LDS once, then sweeps 4 row sub-tiles x 4 K-chunks of WMMA.
// ---------------------------------------------------------------------------
__global__ void gemm_scale_kernel(const float* __restrict__ X, const float* __restrict__ W,
                                  const float* __restrict__ scale, float* __restrict__ Y,
                                  int nrows, const int* __restrict__ mask, int nm) {
  __shared__ _Float16 WsT[DIM * LDW];   // W transposed: [n][k], ~34 KB
  __shared__ _Float16 As[64 * LDW];     // A tile:       [r][k], ~17 KB

  const int tid  = threadIdx.x;
  const int lane = tid & 31;
  const int wave = tid >> 5;            // 0..7
  const int row0 = blockIdx.x * 64;

  // Stage W transposed into LDS as f16 (row stride LDW avoids bank conflicts).
  for (int i = tid; i < DIM * DIM; i += 256) {
    int k = i >> 7, n = i & 127;
    WsT[n * LDW + k] = (_Float16)W[i];
  }
  // Stage A tile; zero rows beyond nrows and masked rows (layer-1 remainder).
  for (int i = tid; i < 64 * DIM; i += 256) {
    int r = i >> 7, col = i & 127;
    int grow = row0 + r;
    float v = 0.0f;
    if (grow < nrows) {
      v = X[(size_t)grow * DIM + col];
      for (int m = 0; m < nm; ++m)
        if (grow == mask[m]) { v = 0.0f; break; }
    }
    As[r * LDW + col] = (_Float16)v;
  }
  __syncthreads();

  const int mrow = lane & 15;                  // A row this lane carries
  const int ncol = (wave << 4) + (lane & 15);  // output column
  const int ahi  = (lane >> 4) * 8;            // A: hi lanes hold K+8 in low quad
  const int bhi  = (lane >> 4) * 8;            // B: hi lanes hold K+8 in low quad
  const int rofs = (lane >> 4) * 8;            // C/D: hi lanes hold M+8

  // B fragments: two 8-half runs at K = kb+bhi and kb+16+bhi -> two 16-B LDS loads each.
  v16h bfrag[4];
#pragma unroll
  for (int kc = 0; kc < 4; ++kc) {
    const _Float16* bp = &WsT[ncol * LDW + kc * 32 + bhi];
    v8h b0 = *(const v8h*)bp;
    v8h b1 = *(const v8h*)(bp + 16);
    bfrag[kc] = __builtin_shufflevector(b0, b1,
        0, 1, 2, 3, 4, 5, 6, 7, 8, 9, 10, 11, 12, 13, 14, 15);
  }

#pragma unroll
  for (int rt = 0; rt < 4; ++rt) {             // 4 row sub-tiles of 16
    v8f acc = {};
#pragma unroll
    for (int kc = 0; kc < 4; ++kc) {           // K = 128 in 4 x 32 steps
      // A fragment: two contiguous 8-half runs at K = kb+ahi and kb+16+ahi.
      const _Float16* ap = &As[(rt * 16 + mrow) * LDW + kc * 32 + ahi];
      v8h a0 = *(const v8h*)ap;
      v8h a1 = *(const v8h*)(ap + 16);
      v16h a = __builtin_shufflevector(a0, a1,
          0, 1, 2, 3, 4, 5, 6, 7, 8, 9, 10, 11, 12, 13, 14, 15);
      acc = __builtin_amdgcn_wmma_f32_16x16x32_f16(false, a, false, bfrag[kc],
                                                   (short)0, acc, false, false);
    }

    // C/D layout: VGPR r <-> M = r + 8*(lane>=16), N = lane&15. A lane's 8
    // output rows are contiguous -> full tiles take a branch-free path with
    // float4 scale loads and immediate-offset stores.
    const int rbase = row0 + rt * 16 + rofs;
    if (rbase + 8 <= nrows) {
      const float4 s0 = *(const float4*)(scale + rbase);
      const float4 s1 = *(const float4*)(scale + rbase + 4);
      float* yp = Y + (size_t)rbase * DIM + ncol;
      yp[0 * DIM] = acc[0] * s0.x;
      yp[1 * DIM] = acc[1] * s0.y;
      yp[2 * DIM] = acc[2] * s0.z;
      yp[3 * DIM] = acc[3] * s0.w;
      yp[4 * DIM] = acc[4] * s1.x;
      yp[5 * DIM] = acc[5] * s1.y;
      yp[6 * DIM] = acc[6] * s1.z;
      yp[7 * DIM] = acc[7] * s1.w;
    } else {
      for (int r = 0; r < 8; ++r) {
        int grow = rbase + r;
        if (grow < nrows) Y[(size_t)grow * DIM + ncol] = acc[r] * scale[grow];
      }
    }
  }
}

// ---------------------------------------------------------------------------
// Edge scatter: AGG[dst[e], :] += H[src[e], :]. 32 threads/edge, float4 wide.
// Both H and AGG are 51 MB -> L2-resident on the 192 MB L2.
// ---------------------------------------------------------------------------
__global__ void scatter_kernel(const float* __restrict__ H, const int* __restrict__ src,
                               const int* __restrict__ dst, float* __restrict__ AGG, int ne) {
  long long idx = (long long)blockIdx.x * 256 + threadIdx.x;
  int e = (int)(idx >> 5);
  if (e >= ne) return;
  int q = ((int)idx & 31) << 2;      // column quad
  int s = src[e], d = dst[e];
  const float4 hv = *(const float4*)(H + (size_t)s * DIM + q);
  float* out = AGG + (size_t)d * DIM + q;
  atomicAdd(out + 0, hv.x);
  atomicAdd(out + 1, hv.y);
  atomicAdd(out + 2, hv.z);
  atomicAdd(out + 3, hv.w);
}

// ---------------------------------------------------------------------------
// y = agg * dn_dst[row] + b  (in place), plus per-column sum / sumsq for BN.
// Block handles 32 rows x 128 cols; block-local reduce then 256 atomics.
// ---------------------------------------------------------------------------
__global__ void bias_stats_kernel(float* __restrict__ Y, const float* __restrict__ dn,
                                  const float* __restrict__ b, float* __restrict__ csum,
                                  float* __restrict__ csq, int nrows) {
  __shared__ float sm[256];
  int t = threadIdx.x;
  int c = t & 127;
  int rh = t >> 7;
  int rbase = blockIdx.x * 32;
  float bc = b[c];
  float s = 0.0f, ss = 0.0f;
  for (int rr = rh; rr < 32; rr += 2) {
    int row = rbase + rr;
    if (row < nrows) {
      size_t ix = (size_t)row * DIM + c;
      float v = Y[ix] * dn[row] + bc;
      Y[ix] = v;
      s += v; ss += v * v;
    }
  }
  sm[t] = s;  __syncthreads();
  if (t < 128) atomicAdd(&csum[c], sm[t] + sm[t + 128]);
  __syncthreads();
  sm[t] = ss; __syncthreads();
  if (t < 128) atomicAdd(&csq[c], sm[t] + sm[t + 128]);
}

// ---------------------------------------------------------------------------
// BN (biased var) + optional PReLU, full matrix.
// ---------------------------------------------------------------------------
__global__ void bn_act_kernel(const float* __restrict__ Y, const float* __restrict__ csum,
                              const float* __restrict__ csq, const float* __restrict__ g,
                              const float* __restrict__ be, const float* __restrict__ ap,
                              float* __restrict__ O, int nrows, int doPrelu) {
  long long i = (long long)blockIdx.x * 256 + threadIdx.x;
  if (i >= (long long)nrows * DIM) return;
  int c = (int)(i & 127);
  float inv_n = 1.0f / (float)nrows;
  float mu  = csum[c] * inv_n;
  float var = csq[c] * inv_n - mu * mu;
  float inv = rsqrtf(var + 1e-5f);
  float v = (Y[i] - mu) * inv * g[c] + be[c];
  if (doPrelu) { float a = *ap; v = (v >= 0.0f) ? v : a * v; }
  O[i] = v;
}

// ---------------------------------------------------------------------------
// Target-branch encoder: 17 nodes, 64 edges — one block, thread = column.
// GraphConv aggregation and BN are column-independent.
// ---------------------------------------------------------------------------
__global__ void target_kernel(const float* __restrict__ feat, const int* __restrict__ mask,
                              const int* __restrict__ ssrc, const int* __restrict__ sdst,
                              const float* __restrict__ W1, const float* __restrict__ b1,
                              const float* __restrict__ g1, const float* __restrict__ be1,
                              const float* __restrict__ a1p,
                              const float* __restrict__ W2, const float* __restrict__ b2,
                              const float* __restrict__ g2, const float* __restrict__ be2,
                              float* __restrict__ vout, int nm, int ne) {
  __shared__ float X[17][DIM], H[17][DIM], Yv[17][DIM];
  __shared__ float dns[17], dnd[17];
  __shared__ int mn[17], es[64], ed[64], cs[17], cd[17];
  if (nm > 17) nm = 17;
  if (ne > 64) ne = 64;
  int c = threadIdx.x;   // 128 threads, one column each

  if (c < nm) { mn[c] = mask[c]; cs[c] = 0; cd[c] = 0; }
  if (c < ne) { es[c] = ssrc[c]; ed[c] = sdst[c]; }
  __syncthreads();
  if (c < ne) { atomicAdd(&cs[es[c]], 1); atomicAdd(&cd[ed[c]], 1); }
  __syncthreads();
  if (c < nm) {
    dns[c] = (cs[c] > 0) ? rsqrtf((float)cs[c]) : 0.0f;
    dnd[c] = (cd[c] > 0) ? rsqrtf((float)cd[c]) : 0.0f;
  }
  for (int r = 0; r < nm; ++r) X[r][c] = feat[(size_t)mn[r] * DIM + c];
  __syncthreads();

  float a1 = *a1p;
  float inv_nm = 1.0f / (float)nm;

  // ---- layer 1: conv + BN + PReLU ----
  for (int r = 0; r < nm; ++r) {
    float acc = 0.0f;
    for (int k = 0; k < DIM; ++k) acc += X[r][k] * W1[k * DIM + c];
    H[r][c] = acc * dns[r];
  }
  for (int r = 0; r < nm; ++r) Yv[r][c] = 0.0f;
  for (int e = 0; e < ne; ++e) Yv[ed[e]][c] += H[es[e]][c];   // own column: no race
  float mu = 0.0f;
  for (int r = 0; r < nm; ++r) { Yv[r][c] = Yv[r][c] * dnd[r] + b1[c]; mu += Yv[r][c]; }
  mu *= inv_nm;
  float var = 0.0f;
  for (int r = 0; r < nm; ++r) { float d0 = Yv[r][c] - mu; var += d0 * d0; }
  var *= inv_nm;
  float inv = rsqrtf(var + 1e-5f);
  for (int r = 0; r < nm; ++r) {
    float v = (Yv[r][c] - mu) * inv * g1[c] + be1[c];
    X[r][c] = (v >= 0.0f) ? v : a1 * v;
  }
  __syncthreads();   // layer-2 matmul reads X across columns

  // ---- layer 2: conv + BN (no activation), then mean over rows ----
  for (int r = 0; r < nm; ++r) {
    float acc = 0.0f;
    for (int k = 0; k < DIM; ++k) acc += X[r][k] * W2[k * DIM + c];
    H[r][c] = acc * dns[r];
  }
  for (int r = 0; r < nm; ++r) Yv[r][c] = 0.0f;
  for (int e = 0; e < ne; ++e) Yv[ed[e]][c] += H[es[e]][c];
  mu = 0.0f;
  for (int r = 0; r < nm; ++r) { Yv[r][c] = Yv[r][c] * dnd[r] + b2[c]; mu += Yv[r][c]; }
  mu *= inv_nm;
  var = 0.0f;
  for (int r = 0; r < nm; ++r) { float d0 = Yv[r][c] - mu; var += d0 * d0; }
  var *= inv_nm;
  inv = rsqrtf(var + 1e-5f);
  float acc = 0.0f;
  for (int r = 0; r < nm; ++r) acc += (Yv[r][c] - mu) * inv * g2[c] + be2[c];
  vout[c] = acc * inv_nm;
}

// ---------------------------------------------------------------------------
// Finalize: apply layer-2 BN to the masked rows, mean them, cosine loss
// against the target vector. One block of 128 threads.
// ---------------------------------------------------------------------------
__global__ void finalize_kernel(const float* __restrict__ Y2, const int* __restrict__ mask,
                                const float* __restrict__ csum, const float* __restrict__ csq,
                                const float* __restrict__ g, const float* __restrict__ be,
                                const float* __restrict__ vtgt, float* __restrict__ out,
                                int nrows, int nm) {
  __shared__ float red0[128], red1[128], red2[128];
  int c = threadIdx.x;   // 128 threads
  float inv_n = 1.0f / (float)nrows;
  float mu  = csum[c] * inv_n;
  float var = csq[c] * inv_n - mu * mu;
  float inv = rsqrtf(var + 1e-5f);
  float u = 0.0f;
  for (int r = 0; r < nm; ++r) {
    float y = Y2[(size_t)mask[r] * DIM + c];
    u += (y - mu) * inv * g[c] + be[c];
  }
  u *= 1.0f / (float)nm;
  float v = vtgt[c];
  red0[c] = u * v; red1[c] = u * u; red2[c] = v * v;
  __syncthreads();
  for (int s = 64; s > 0; s >>= 1) {
    if (c < s) { red0[c] += red0[c + s]; red1[c] += red1[c + s]; red2[c] += red2[c + s]; }
    __syncthreads();
  }
  if (c == 0) {
    float nx = fmaxf(sqrtf(red1[0]), 1e-12f);
    float ny = fmaxf(sqrtf(red2[0]), 1e-12f);
    out[0] = 1.0f - red0[0] / (nx * ny);
  }
}

// ---------------------------------------------------------------------------
extern "C" void kernel_launch(void* const* d_in, const int* in_sizes, int n_in,
                              void* d_out, int out_size, void* d_ws, size_t ws_size,
                              hipStream_t stream) {
  (void)n_in; (void)out_size; (void)ws_size;

  const float* feat = (const float*)d_in[0];
  const int*   src  = (const int*)d_in[1];
  const int*   dst  = (const int*)d_in[2];
  const int*   ssrc = (const int*)d_in[3];
  const int*   sdst = (const int*)d_in[4];
  const int*   mask = (const int*)d_in[5];
  const float* W1o  = (const float*)d_in[6];
  const float* b1o  = (const float*)d_in[7];
  const float* g1o  = (const float*)d_in[8];
  const float* be1o = (const float*)d_in[9];
  const float* a1o  = (const float*)d_in[10];
  const float* W2o  = (const float*)d_in[11];
  const float* b2o  = (const float*)d_in[12];
  const float* g2o  = (const float*)d_in[13];
  const float* be2o = (const float*)d_in[14];
  const float* W1t  = (const float*)d_in[16];
  const float* b1t  = (const float*)d_in[17];
  const float* g1t  = (const float*)d_in[18];
  const float* be1t = (const float*)d_in[19];
  const float* a1t  = (const float*)d_in[20];
  const float* W2t  = (const float*)d_in[21];
  const float* b2t  = (const float*)d_in[22];
  const float* g2t  = (const float*)d_in[23];
  const float* be2t = (const float*)d_in[24];

  const int N_  = in_sizes[0] / DIM;   // 100000 nodes
  const int E_  = in_sizes[1];         // 1.6M edges
  const int ES_ = in_sizes[3];         // 64 subgraph edges
  const int NM_ = in_sizes[5];         // 17 masked nodes

  // ---- workspace carve-up ----
  float* p = (float*)d_ws;
  float* dn_src = p; p += N_;
  float* dn_dst = p; p += N_;
  float* cs1 = p; p += 128;
  float* cq1 = p; p += 128;
  float* cs2 = p; p += 128;
  float* cq2 = p; p += 128;
  float* vtgt = p; p += 128;
  uintptr_t up = (uintptr_t)p;
  up = (up + 255) & ~(uintptr_t)255;
  p = (float*)up;
  float* bufA = p; p += (size_t)N_ * DIM;
  float* bufB = p; // p += (size_t)N_ * DIM;

  // zero degrees + all column stats + target vec in one contiguous memset
  hipMemsetAsync(dn_src, 0, sizeof(float) * ((size_t)2 * N_ + 5 * 128), stream);

  const int eb = (E_ + 255) / 256;
  const int nb = (N_ + 255) / 256;
  const int gb = (N_ + 63) / 64;
  const int sb = (int)(((long long)E_ * 32 + 255) / 256);
  const int rb = (N_ + 31) / 32;
  const int fb = (int)(((long long)N_ * DIM + 255) / 256);
  const size_t matBytes = (size_t)N_ * DIM * sizeof(float);

  // degrees
  deg_count_kernel<<<eb, 256, 0, stream>>>(src, dst, dn_src, dn_dst, E_);
  deg_finish_kernel<<<nb, 256, 0, stream>>>(dn_src, dn_dst, N_);

  // ---- online layer 1 ----
  gemm_scale_kernel<<<gb, 256, 0, stream>>>(feat, W1o, dn_src, bufA, N_, mask, NM_);
  hipMemsetAsync(bufB, 0, matBytes, stream);
  scatter_kernel<<<sb, 256, 0, stream>>>(bufA, src, dst, bufB, E_);
  bias_stats_kernel<<<rb, 256, 0, stream>>>(bufB, dn_dst, b1o, cs1, cq1, N_);
  bn_act_kernel<<<fb, 256, 0, stream>>>(bufB, cs1, cq1, g1o, be1o, a1o, bufA, N_, 1);

  // ---- online layer 2 (BN application deferred to finalize; stats only) ----
  gemm_scale_kernel<<<gb, 256, 0, stream>>>(bufA, W2o, dn_src, bufB, N_, nullptr, 0);
  hipMemsetAsync(bufA, 0, matBytes, stream);
  scatter_kernel<<<sb, 256, 0, stream>>>(bufB, src, dst, bufA, E_);
  bias_stats_kernel<<<rb, 256, 0, stream>>>(bufA, dn_dst, b2o, cs2, cq2, N_);

  // ---- target branch + loss ----
  target_kernel<<<1, 128, 0, stream>>>(feat, mask, ssrc, sdst,
                                       W1t, b1t, g1t, be1t, a1t,
                                       W2t, b2t, g2t, be2t, vtgt, NM_, ES_);
  finalize_kernel<<<1, 128, 0, stream>>>(bufA, mask, cs2, cq2, g2o, be2o, vtgt,
                                         (float*)d_out, N_, NM_);
}
